// AttentionHead_2654289789070
// MI455X (gfx1250) — matrix operand order repaired and checked by reading, not compiled, not run on hardware
//
#include <hip/hip_runtime.h>
#include <hip/hip_bf16.h>

typedef __attribute__((ext_vector_type(16))) __bf16         v16bf;
typedef __attribute__((ext_vector_type(2)))  __bf16         v2bf;
typedef __attribute__((ext_vector_type(8)))  float          v8f;
typedef __attribute__((ext_vector_type(4)))  float          v4f;
typedef __attribute__((ext_vector_type(8)))  unsigned int   v8u;
typedef __attribute__((ext_vector_type(4)))  unsigned int   v4u;

#define BATCH   4
#define T1      4096
#define T2      4096
#define DIN     512
#define EMB     64
#define MINF    (-1000000000.0f)
// 1/sqrt(64) * log2(e): scores pre-scaled into the exp2 domain
#define SCALE2  0.18033688011112042f

// fp32 -> bf16 round-to-nearest-even (scalar fallback)
__device__ __forceinline__ unsigned short f2bf(float f) {
    unsigned int u = __builtin_bit_cast(unsigned int, f);
    u += 0x7FFFu + ((u >> 16) & 1u);
    return (unsigned short)(u >> 16);
}

// pack two fp32 -> dword of 2 bf16 (a -> low 16, b -> high 16)
__device__ __forceinline__ unsigned int f2bf2(float a, float b) {
#if __has_builtin(__builtin_amdgcn_cvt_pk_bf16_f32)
    v2bf r = __builtin_amdgcn_cvt_pk_bf16_f32(a, b);
    return __builtin_bit_cast(unsigned int, r);
#else
    return (unsigned int)f2bf(a) | ((unsigned int)f2bf(b) << 16);
#endif
}

__device__ __forceinline__ v8f vzero8() {
    v8f z = {0.f,0.f,0.f,0.f,0.f,0.f,0.f,0.f};
    return z;
}

struct dpair { v4u lo, hi; };   // 32 B, matches v16bf byte layout

__device__ __forceinline__ v16bf pack2(v4u lo, v4u hi) {
    dpair p{lo, hi};
    return __builtin_bit_cast(v16bf, p);
}

// -------------------------------------------------------------------------
// Kernel 0: pack W[512x64] fp32 into bf16 WMMA B-operand tiles.
// Lane layout used here (contiguous k, same as A): n = lane%16, element i <-> k = 16*(lane/16)+i.
// Stored as 32 B per lane -> 1 KB per tile, [mat][s][j][lane][16].
// -------------------------------------------------------------------------
__global__ void pack_w(const float* __restrict__ Wq,
                       const float* __restrict__ Wk,
                       const float* __restrict__ Wv,
                       unsigned short* __restrict__ wpack) {
    const int s   = blockIdx.x;          // 0..15 kstep
    const int j   = blockIdx.y;          // 0..3  ntile
    const int mat = blockIdx.z;          // 0..2
    const float* W = (mat == 0) ? Wq : (mat == 1) ? Wk : Wv;
    const int lane = threadIdx.x & 31;
    const int n  = j * 16 + (lane & 15);
    const int kb = s * 32 + ((lane >> 4) << 4);
    v8u vals;
#pragma unroll
    for (int t = 0; t < 8; ++t)
        vals[t] = f2bf2(W[(kb + 2 * t) * EMB + n], W[(kb + 2 * t + 1) * EMB + n]);
    unsigned short* dst = wpack + (((mat * 16 + s) * 4 + j) << 9) + lane * 16;
    *reinterpret_cast<v8u*>(dst) = vals;
}

// -------------------------------------------------------------------------
// Kernel 1: projection  out[m, e] = X[m, :512] . W[:, e] + bias[e]  (bf16 out)
// One wave = one 16-row M tile x N=64.  TR=1 stores Vt[b][e][t].
// -------------------------------------------------------------------------
template <int TR>
__global__ void project(const float* __restrict__ X,
                        const unsigned short* __restrict__ wt,
                        const float* __restrict__ bias,
                        unsigned short* __restrict__ out) {
    const int wave  = threadIdx.x >> 5;
    const int lane  = threadIdx.x & 31;
    const int l15   = lane & 15;
    const int g     = lane >> 4;
    const int mbase = (blockIdx.x * 4 + wave) * 16;
    const int mrow  = mbase + l15;

    v8f acc[4];
#pragma unroll
    for (int j = 0; j < 4; ++j) acc[j] = vzero8();

    for (int s = 0; s < 16; ++s) {
        const float* xr = X + mrow * DIN + 32 * s + 16 * g;
        v4f a0 = *reinterpret_cast<const v4f*>(xr);
        v4f a1 = *reinterpret_cast<const v4f*>(xr + 4);
        v4f a2 = *reinterpret_cast<const v4f*>(xr + 8);
        v4f a3 = *reinterpret_cast<const v4f*>(xr + 12);
        v8u au;
#pragma unroll
        for (int t = 0; t < 2; ++t) {
            au[t]     = f2bf2(a0[2 * t], a0[2 * t + 1]);
            au[2 + t] = f2bf2(a1[2 * t], a1[2 * t + 1]);
            au[4 + t] = f2bf2(a2[2 * t], a2[2 * t + 1]);
            au[6 + t] = f2bf2(a3[2 * t], a3[2 * t + 1]);
        }
        v16bf A = __builtin_bit_cast(v16bf, au);
#pragma unroll
        for (int j = 0; j < 4; ++j) {
            v8u bu = *reinterpret_cast<const v8u*>(wt + ((s * 4 + j) << 9) + lane * 16);
            acc[j] = __builtin_amdgcn_wmma_f32_16x16x32_bf16(
                false, A, false, __builtin_bit_cast(v16bf, bu), (short)0, acc[j], false, false);
        }
    }

#pragma unroll
    for (int j = 0; j < 4; ++j) {
        const int e = 16 * j + l15;
        const float bv = bias[e];
#pragma unroll
        for (int r = 0; r < 8; ++r) {
            const unsigned short val = f2bf(acc[j][r] + bv);
            const int m = mbase + r + 8 * g;
            if (TR == 0) {
                out[m * EMB + e] = val;
            } else {
                out[((m >> 12) * EMB + e) * T2 + (m & (T2 - 1))] = val;
            }
        }
    }
}

// -------------------------------------------------------------------------
// Kernel 2: flash attention. One wave = 16 query rows, chunks of 32 keys.
// Row max via 16-lane butterflies; row SUM via WMMA against all-ones B tile
// (same D-layout as O accumulators, same corr rescale -> no sum shuffles).
// -------------------------------------------------------------------------
__global__ void attn(const unsigned short* __restrict__ Qb,   // [B*T1, 64] bf16
                     const unsigned short* __restrict__ Kb,   // [B*T2, 64] bf16
                     const unsigned short* __restrict__ Vtb,  // [B][64][T2] bf16
                     const float* __restrict__ mask,          // [B, T2]
                     float* __restrict__ out) {               // [B*T1, 64] f32
    __shared__ unsigned short pstage[4 * 16 * 32];            // 1 KB per wave

    const int wave  = threadIdx.x >> 5;
    const int lane  = threadIdx.x & 31;
    const int l15   = lane & 15;
    const int g     = lane >> 4;
    const int mbase = (blockIdx.x * 4 + wave) * 16;           // global query row
    const int b     = mbase >> 12;                            // / T1

    // Q A-tiles (embed steps 0,1) resident in registers
    v16bf qa[2];
    const unsigned short* qrow = Qb + (mbase + l15) * EMB;
#pragma unroll
    for (int es = 0; es < 2; ++es) {
        v4u lo = *reinterpret_cast<const v4u*>(qrow + 32 * es + 16 * g);
        v4u hi = *reinterpret_cast<const v4u*>(qrow + 32 * es + 8 + 16 * g);
        qa[es] = pack2(lo, hi);
    }

    // all-ones bf16 B tile (1.0bf16 = 0x3F80) for row sums of P
    v8u onesu;
#pragma unroll
    for (int t = 0; t < 8; ++t) onesu[t] = 0x3F803F80u;
    const v16bf ones = __builtin_bit_cast(v16bf, onesu);

    const unsigned short* K  = Kb  + b * T2 * EMB;
    const unsigned short* Vt = Vtb + b * EMB * T2;
    const float*          mk = mask + b * T2;
    unsigned short*       st = pstage + wave * 512;           // [16][32]

    float rmax[8];
    v8f o[4], osum;
#pragma unroll
    for (int r = 0; r < 8; ++r) rmax[r] = -3.0e38f;
#pragma unroll
    for (int j = 0; j < 4; ++j) o[j] = vzero8();
    osum = vzero8();

    for (int t2 = 0; t2 < T2; t2 += 32) {
        if (t2 + 32 < T2) {
            __builtin_prefetch(K + (t2 + 32 + l15) * EMB, 0, 1);
            __builtin_prefetch(Vt + l15 * T2 + t2 + 32, 0, 1);
        }
        // ---- S = Q . K^T  (two 16-wide n tiles)
        v8f s0 = vzero8(), s1 = vzero8();
#pragma unroll
        for (int es = 0; es < 2; ++es) {
            v8u b0 = *reinterpret_cast<const v8u*>(
                K + (t2 + l15) * EMB + 32 * es + 16 * g);
            v8u b1 = *reinterpret_cast<const v8u*>(
                K + (t2 + 16 + l15) * EMB + 32 * es + 16 * g);
            s0 = __builtin_amdgcn_wmma_f32_16x16x32_bf16(
                false, qa[es], false, __builtin_bit_cast(v16bf, b0), (short)0, s0, false, false);
            s1 = __builtin_amdgcn_wmma_f32_16x16x32_bf16(
                false, qa[es], false, __builtin_bit_cast(v16bf, b1), (short)0, s1, false, false);
        }
        // ---- scale into exp2 domain + mask (0/1 mask -> value or MINF)
        const float m0 = mk[t2 + l15];
        const float m1 = mk[t2 + 16 + l15];
#pragma unroll
        for (int r = 0; r < 8; ++r) {
            s0[r] = (m0 > 0.5f) ? s0[r] * SCALE2 : MINF;
            s1[r] = (m1 > 0.5f) ? s1[r] * SCALE2 : MINF;
        }
        // ---- chunk row max across 16 n-lanes
        float cm[8];
#pragma unroll
        for (int r = 0; r < 8; ++r) cm[r] = fmaxf(s0[r], s1[r]);
#pragma unroll
        for (int off = 1; off <= 8; off <<= 1)
#pragma unroll
            for (int r = 0; r < 8; ++r)
                cm[r] = fmaxf(cm[r], __shfl_xor(cm[r], off, 32));
        // ---- online rescale + P -> LDS (bf16), no live P arrays
#pragma unroll
        for (int r = 0; r < 8; ++r) {
            const float nm   = fmaxf(rmax[r], cm[r]);
            const float corr = exp2f(rmax[r] - nm);
            rmax[r] = nm;
            osum[r] *= corr;
#pragma unroll
            for (int j = 0; j < 4; ++j) o[j][r] *= corr;
            const int m = r + 8 * g;
            st[m * 32 + l15]      = f2bf(exp2f(s0[r] - nm));
            st[m * 32 + 16 + l15] = f2bf(exp2f(s1[r] - nm));
        }
        // ---- reload P in A-layout, accumulate O += P.V and rowsum += P.1
        {
            const unsigned short* sp = st + l15 * 32;
            v4u lo = *reinterpret_cast<const v4u*>(sp + 16 * g);
            v4u hi = *reinterpret_cast<const v4u*>(sp + 8 + 16 * g);
            v16bf pa = pack2(lo, hi);
#pragma unroll
            for (int j = 0; j < 4; ++j) {
                v8u vv = *reinterpret_cast<const v8u*>(
                    Vt + (16 * j + l15) * T2 + t2 + 16 * g);
                o[j] = __builtin_amdgcn_wmma_f32_16x16x32_bf16(
                    false, pa, false, __builtin_bit_cast(v16bf, vv), (short)0, o[j], false, false);
            }
            osum = __builtin_amdgcn_wmma_f32_16x16x32_bf16(
                false, pa, false, ones, (short)0, osum, false, false);
        }
    }

    // ---- epilogue: normalize and store fp32
#pragma unroll
    for (int r = 0; r < 8; ++r) {
        const float inv = 1.0f / osum[r];
        const int m = mbase + r + 8 * g;
#pragma unroll
        for (int j = 0; j < 4; ++j)
            out[m * EMB + 16 * j + l15] = o[j][r] * inv;
    }
}

// -------------------------------------------------------------------------
extern "C" void kernel_launch(void* const* d_in, const int* in_sizes, int n_in,
                              void* d_out, int out_size, void* d_ws, size_t ws_size,
                              hipStream_t stream) {
    const float* q    = (const float*)d_in[0];
    const float* k    = (const float*)d_in[1];
    const float* v    = (const float*)d_in[2];
    const float* mask = (const float*)d_in[3];
    const float* Wq   = (const float*)d_in[4];
    const float* bq   = (const float*)d_in[5];
    const float* Wk   = (const float*)d_in[6];
    const float* bk   = (const float*)d_in[7];
    const float* Wv   = (const float*)d_in[8];
    const float* bv   = (const float*)d_in[9];

    char* ws = (char*)d_ws;
    const size_t WPACK_BYTES = 3u * 16u * 4u * 1024u;         // 192 KB
    const size_t MAT_BYTES   = (size_t)BATCH * T1 * EMB * 2u; // 2 MB each
    unsigned short* wpack = (unsigned short*)ws;
    unsigned short* Qb    = (unsigned short*)(ws + WPACK_BYTES);
    unsigned short* Kb    = (unsigned short*)(ws + WPACK_BYTES + MAT_BYTES);
    unsigned short* Vt    = (unsigned short*)(ws + WPACK_BYTES + 2 * MAT_BYTES);
    const unsigned W_MAT_USHORTS = 16u * 4u * 512u;           // ushorts per matrix

    pack_w<<<dim3(16, 4, 3), 32, 0, stream>>>(Wq, Wk, Wv, wpack);

    const int mblocks = (BATCH * T1 / 16) / 4;                // 256
    project<0><<<mblocks, 128, 0, stream>>>(q, wpack + 0 * W_MAT_USHORTS, bq, Qb);
    project<0><<<mblocks, 128, 0, stream>>>(k, wpack + 1 * W_MAT_USHORTS, bk, Kb);
    project<1><<<mblocks, 128, 0, stream>>>(v, wpack + 2 * W_MAT_USHORTS, bv, Vt);

    attn<<<mblocks, 128, 0, stream>>>(Qb, Kb, Vt, mask, (float*)d_out);

    (void)in_sizes; (void)n_in; (void)out_size; (void)ws_size;
}
